// SelfAttentionBlock_21234318312246
// MI455X (gfx1250) — matrix operand order repaired and checked by reading, not compiled, not run on hardware
//
#include <hip/hip_runtime.h>
#include <stdint.h>

typedef __attribute__((ext_vector_type(16))) __bf16 v16bf;
typedef __attribute__((ext_vector_type(8)))  float  v8f;

// ws layout: [0, 32768) bf16  : W_Q fragments (16 ntiles x 2 ktiles x 32 lanes x 16 elems)
//            [32768,65536) bf16: W_K fragments (same shape)
//            [65536, 66560) f32: W_V (4 x 8 x 8)

__global__ void __launch_bounds__(256)
prep_kernel(const float* __restrict__ WQ, const float* __restrict__ WK,
            const float* __restrict__ bases, const float* __restrict__ coeffs,
            __bf16* __restrict__ wfrag, float* __restrict__ wv) {
  int idx = blockIdx.x * 256 + threadIdx.x;
  if (idx < 32768) {
    int wsel = idx >> 14;          // 0 = W_Q, 1 = W_K
    int r    = idx & 16383;
    int nt   = r >> 10;            // N tile (0..15)
    int kt   = (r >> 9) & 1;       // K tile (0..1)
    int lane = (r >> 4) & 31;
    int e    = r & 15;
    // B-matrix (32x16, bf16) lane layout: lanes 0-15 hold K=0..15 (N=lane),
    // lanes 16-31 hold K=16..31 (N=lane-16)  [contiguous k order, same as the A fragments]
    int col = lane & 15;
    int kl  = (lane < 16) ? e : (e + 16);
    int K   = kt * 32 + kl;
    const float* W = wsel ? WK : WQ;      // W is [256,64]; B[k][n] = W[n*64+k]
    wfrag[idx] = (__bf16)W[(nt * 16 + col) * 64 + K];
  } else if (idx < 32768 + 256) {
    int t = idx - 32768;
    int h = t >> 6;
    int ij = t & 63;
    float acc = 0.f;
    #pragma unroll
    for (int b = 0; b < 6; ++b) acc += coeffs[h * 6 + b] * bases[b * 64 + ij];
    wv[t] = acc;
  }
}

__global__ void __launch_bounds__(32)
attn_kernel(const float* __restrict__ x, const long long* __restrict__ nbr,
            const float* __restrict__ ptm,
            const float* __restrict__ qb, const float* __restrict__ kb,
            const __bf16* __restrict__ wfrag, const float* __restrict__ wv,
            float* __restrict__ out) {
  const int v = blockIdx.x;
  const int l = threadIdx.x;

  __shared__ float s_x[64];          // this vertex's features
  __shared__ float s_q[256];         // Q
  __shared__ float s_qb[256], s_kb[256], s_wv[256];
  __shared__ float s_xn[16][64];     // all neighbor features   (async staged)
  __shared__ float s_P[16][64];      // all transport matrices  (async staged)
  __shared__ float s_fp[16][64];     // f_prime_flat per neighbor (fp32)
  __shared__ float s_K[16][256];     // K per neighbor
  __shared__ float s_sc[16][4], s_at[16][4];
  __shared__ float s_G[4][64];       // attn-weighted f_prime per head

  // ---- async-stage ALL neighbor data straight into LDS (no VGPR round trip).
  // One GLOBAL_LOAD_ASYNC_TO_LDS_B128 per neighbor: lanes 0-15 move the 64-float
  // x-row, lanes 16-31 move the 64-float P matrix (16B per lane).
  {
    const int q16 = l & 15;                // 16B chunk id within the 256B row
    #pragma unroll
    for (int n = 0; n < 16; ++n) {
      long long ni = nbr[(size_t)v * 16 + n];
      const float* src = (l < 16) ? (x   + (size_t)ni * 64        + 4 * q16)
                                  : (ptm + ((size_t)v * 16 + n) * 64 + 4 * q16);
      float* dstp = (l < 16) ? &s_xn[n][4 * q16] : &s_P[n][4 * q16];
      uint32_t lds_off = (uint32_t)(uintptr_t)dstp;   // LDS aperture: addr[31:0]
      uint64_t gaddr   = (uint64_t)(uintptr_t)src;
      asm volatile("global_load_async_to_lds_b128 %0, %1, off"
                   :: "v"(lds_off), "v"(gaddr) : "memory");
    }
  }

  s_x[l]      = x[(size_t)v * 64 + l];
  s_x[l + 32] = x[(size_t)v * 64 + 32 + l];
  #pragma unroll
  for (int i = 0; i < 8; ++i) {
    s_qb[l * 8 + i] = qb[l * 8 + i];
    s_kb[l * 8 + i] = kb[l * 8 + i];
    s_wv[l * 8 + i] = wv[l * 8 + i];
  }

  asm volatile("s_wait_asynccnt 0x0" ::: "memory");
  __syncthreads();

  // ---- parallel transport: f_prime[n] = P[n] (8x8) @ Xn^T (8x8), VALU ----
  #pragma unroll 2
  for (int n = 0; n < 16; ++n) {
    #pragma unroll
    for (int half = 0; half < 2; ++half) {
      int k = l + 32 * half;            // k = o*8 + p (row-major [N,C] flatten)
      int o = k >> 3, p = k & 7;
      float acc = 0.f;
      #pragma unroll
      for (int j = 0; j < 8; ++j) acc += s_P[n][o * 8 + j] * s_xn[n][p * 8 + j];
      s_fp[n][k] = acc;
    }
  }
  __syncthreads();

  // ---- build bf16 A fragments (16x32, contiguous k order, same as B fragments) ----
  // lane<16: row=lane, K(e) = e ; lane>=16: row=lane-16, K(e) = e+16
  v16bf aK[2], aQ[2];
  const int arow = l & 15;
  #pragma unroll
  for (int kt = 0; kt < 2; ++kt) {
    #pragma unroll
    for (int e = 0; e < 16; ++e) {
      int kl = (l < 16) ? e : (e + 16);
      int K  = kt * 32 + kl;
      aK[kt][e] = (__bf16)s_fp[arow][K];   // K projection: rows = neighbors
      aQ[kt][e] = (__bf16)s_x[K];          // Q: x_v replicated across all 16 rows
    }
  }

  // ---- WMMA: K = f' @ W_K^T (16x256) and Q = x @ W_Q^T (row-replicated) ----
  const __bf16* bQ = wfrag;
  const __bf16* bK = wfrag + 16384;
  const int colid = l & 15;
  for (int nt = 0; nt < 16; ++nt) {
    v16bf bq0 = *(const v16bf*)(bQ + nt * 1024 +       l * 16);
    v16bf bq1 = *(const v16bf*)(bQ + nt * 1024 + 512 + l * 16);
    v16bf bk0 = *(const v16bf*)(bK + nt * 1024 +       l * 16);
    v16bf bk1 = *(const v16bf*)(bK + nt * 1024 + 512 + l * 16);
    v8f ck = {};
    ck = __builtin_amdgcn_wmma_f32_16x16x32_bf16(false, aK[0], false, bk0, (short)0, ck, false, false);
    ck = __builtin_amdgcn_wmma_f32_16x16x32_bf16(false, aK[1], false, bk1, (short)0, ck, false, false);
    v8f cq = {};
    cq = __builtin_amdgcn_wmma_f32_16x16x32_bf16(false, aQ[0], false, bq0, (short)0, cq, false, false);
    cq = __builtin_amdgcn_wmma_f32_16x16x32_bf16(false, aQ[1], false, bq1, (short)0, cq, false, false);
    int col = nt * 16 + colid;
    #pragma unroll
    for (int r = 0; r < 8; ++r) {            // C/D layout: VGPR r -> M = r (+8 for hi lanes)
      int row = (l < 16) ? r : (r + 8);
      s_K[row][col] = ck[r] + s_kb[col];
    }
    if (l < 16) s_q[col] = cq[0] + s_qb[col];  // any row; all rows identical
  }
  __syncthreads();

  // ---- scores + softmax over neighbors ----
  const int sn = l & 15;
  const int hb = l >> 4;                      // 0 or 1
  #pragma unroll
  for (int hh = 0; hh < 2; ++hh) {
    int h = hb + 2 * hh;
    float acc = 0.f;
    for (int d = 0; d < 64; ++d) acc += s_q[h * 64 + d] * s_K[sn][h * 64 + d];
    s_sc[sn][h] = acc * 0.125f;               // 1/sqrt(64)
  }
  __syncthreads();
  #pragma unroll
  for (int hh = 0; hh < 2; ++hh) {
    int h = hb + 2 * hh;
    float m = -1e30f;
    for (int nn = 0; nn < 16; ++nn) m = fmaxf(m, s_sc[nn][h]);
    float den = 0.f;
    for (int nn = 0; nn < 16; ++nn) den += __expf(s_sc[nn][h] - m);
    s_at[sn][h] = __expf(s_sc[sn][h] - m) / den;
  }
  __syncthreads();

  // ---- G[h][k] = sum_n attn[n,h] * f'[n][k]  (refactored value stage) ----
  const int gh = l >> 3;
  const int gk = (l & 7) * 8;
  #pragma unroll
  for (int t = 0; t < 8; ++t) {
    float acc = 0.f;
    for (int n = 0; n < 16; ++n) acc += s_at[n][gh] * s_fp[n][gk + t];
    s_G[gh][gk + t] = acc;
  }
  __syncthreads();

  // ---- out[h, c*8+i] = sum_j W_V[h,i,j] * G[h][c*8+j] ----
  const int oh = l >> 3, oc = l & 7;
  #pragma unroll
  for (int i = 0; i < 8; ++i) {
    float acc = 0.f;
    #pragma unroll
    for (int j = 0; j < 8; ++j)
      acc += s_wv[oh * 64 + i * 8 + j] * s_G[oh][oc * 8 + j];
    out[(size_t)v * 256 + oh * 64 + oc * 8 + i] = acc;
  }
}

extern "C" void kernel_launch(void* const* d_in, const int* in_sizes, int n_in,
                              void* d_out, int out_size, void* d_ws, size_t ws_size,
                              hipStream_t stream) {
  const float*     x      = (const float*)d_in[0];
  const long long* nbr    = (const long long*)d_in[1];   // int64
  const float*     ptm    = (const float*)d_in[2];
  /* d_in[3] = rel_pos_u: unused by reference */
  const float*     WQw    = (const float*)d_in[4];
  const float*     WQb    = (const float*)d_in[5];
  const float*     WKw    = (const float*)d_in[6];
  const float*     WKb    = (const float*)d_in[7];
  const float*     bases  = (const float*)d_in[8];
  const float*     coeffs = (const float*)d_in[9];
  float* out = (float*)d_out;
  const int V = in_sizes[0] / 64;

  __bf16* wfrag = (__bf16*)d_ws;                          // 64 KB
  float*  wv    = (float*)((char*)d_ws + 65536);          // 1 KB

  prep_kernel<<<129, 256, 0, stream>>>(WQw, WKw, bases, coeffs, wfrag, wv);
  attn_kernel<<<V, 32, 0, stream>>>(x, nbr, ptm, WQb, WKb, wfrag, wv, out);
}
